// TransformerBlock_2293512536235
// MI455X (gfx1250) — matrix operand order repaired and checked by reading, not compiled, not run on hardware
//
#include <hip/hip_runtime.h>
#include <cstdint>

// ---------------- problem constants ----------------
#define BB   4
#define SS   2048
#define EE   1024
#define HH   16
#define DD   64
#define FF_  4096
#define MM   (BB * SS)   // 8192 rows of the activation matrix

// ---------------- WMMA types ----------------
typedef __attribute__((ext_vector_type(8)))  __bf16 v8bf;
typedef __attribute__((ext_vector_type(16))) __bf16 v16bf;
typedef __attribute__((ext_vector_type(8)))  float  v8f;

// ---------------- optional CDNA5 async global->LDS path ----------------
#define AS1 __attribute__((address_space(1)))
#define AS3 __attribute__((address_space(3)))
typedef int i32x4 __attribute__((vector_size(16)));
#if defined(__gfx1250__) && __has_builtin(__builtin_amdgcn_global_load_async_to_lds_b128)
#define USE_ASYNC_LDS 1
__device__ __forceinline__ void async_copy16(const __bf16* gp, __bf16* lp) {
  // 16 bytes per lane, memory -> LDS, tracked by ASYNCcnt
  __builtin_amdgcn_global_load_async_to_lds_b128(
      (AS1 i32x4*)(uintptr_t)gp, (AS3 i32x4*)(uintptr_t)lp, 0, 0);
}
__device__ __forceinline__ void async_wait0() {
#if __has_builtin(__builtin_amdgcn_s_wait_asynccnt)
  __builtin_amdgcn_s_wait_asynccnt(0);
#else
  asm volatile("s_wait_asynccnt 0x0" ::: "memory");
#endif
}
#endif

__device__ __forceinline__ v8f wmma_bf16(v16bf a, v16bf b, v8f c) {
  // (neg_a, A, neg_b, B, c_mod, C, reuse_a, reuse_b)
  return __builtin_amdgcn_wmma_f32_16x16x32_bf16(false, a, false, b, (short)0, c,
                                                 false, false);
}

// 16 contiguous bf16 (16B-aligned) -> one fragment register set
__device__ __forceinline__ v16bf ld16(const __bf16* p) {
  union { v16bf v; v8bf h[2]; } u;
  u.h[0] = *(const v8bf*)p;
  u.h[1] = *(const v8bf*)(p + 8);
  return u.v;
}
// two separate 8-element chunks (A-matrix layout)
__device__ __forceinline__ v16bf ld8x2(const __bf16* lo, const __bf16* hi) {
  union { v16bf v; v8bf h[2]; } u;
  u.h[0] = *(const v8bf*)lo;
  u.h[1] = *(const v8bf*)hi;
  return u.v;
}

// ---------------- fp32 -> bf16 conversion ----------------
__global__ void cvt_bf16_kernel(const float* __restrict__ in,
                                __bf16* __restrict__ out, int n) {
  int i = blockIdx.x * blockDim.x + threadIdx.x;
  int stride = gridDim.x * blockDim.x;
  for (; i < n; i += stride) out[i] = (__bf16)in[i];
}

// ---------------- generic bf16 WMMA GEMM ----------------
// C[M,N] = act(A[M,K] @ B[K,N] + bias) (+ resid); writes f32 and/or bf16.
// block = 256 threads (8 waves), tile 128x128, K-step 32.
__global__ __launch_bounds__(256)
void gemm_bf16_kernel(const __bf16* __restrict__ A, const __bf16* __restrict__ B,
                      const float* __restrict__ bias, const float* __restrict__ resid,
                      float* __restrict__ Cf, __bf16* __restrict__ Cb,
                      int M, int N, int K, int relu) {
  __shared__ __align__(16) __bf16 As[128][40];   // [m][k]   (pad: 80B stride)
  __shared__ __align__(16) __bf16 Bs[128][40];   // [n][k]   transposed stage

  const int tid  = threadIdx.x;
  const int lane = tid & 31;
  const int ln   = lane & 15;        // column / row-within-16 index
  const int lh   = lane >> 4;        // wave half
  const int w    = tid >> 5;
  const int wm   = w >> 2;           // 0..1  (64 rows each)
  const int wn   = w & 3;            // 0..3  (32 cols each)
  const int mb   = blockIdx.y * 128;
  const int nb   = blockIdx.x * 128;

  v8f acc[4][2];
  const v8f vzero = {0.f, 0.f, 0.f, 0.f, 0.f, 0.f, 0.f, 0.f};
#pragma unroll
  for (int mi = 0; mi < 4; ++mi)
#pragma unroll
    for (int ni = 0; ni < 2; ++ni) acc[mi][ni] = vzero;

  for (int kt = 0; kt < K; kt += 32) {
    __syncthreads();
#ifdef USE_ASYNC_LDS
    // stage A tile 128x32 via async copy (2 x b128 per lane, ASYNCcnt)
#pragma unroll
    for (int i = 0; i < 2; ++i) {
      int c = tid + i * 256;          // 512 chunks of 16B
      int r = c >> 2;                 // 4 chunks per 64B row
      int e = (c & 3) << 3;           // element offset 0,8,16,24
      async_copy16(A + (size_t)(mb + r) * K + kt + e, &As[r][e]);
    }
#else
    if (kt + 32 < K) {  // prefetch next K-tile (global_prefetch_b8)
      __builtin_prefetch(A + (size_t)(mb + (tid >> 4)) * K + (kt + 32) + ((tid & 15) << 1), 0, 1);
      __builtin_prefetch(B + (size_t)(kt + 32 + (tid >> 6)) * N + nb + ((tid & 63) << 1), 0, 1);
    }
    // stage A tile 128x32 (8 dwords/thread, coalesced along K)
#pragma unroll
    for (int i = 0; i < 8; ++i) {
      int idx = tid + i * 256;
      int r  = idx >> 4;
      int kc = (idx & 15) << 1;
      uint32_t d = *(const uint32_t*)(A + (size_t)(mb + r) * K + kt + kc);
      *(uint32_t*)&As[r][kc] = d;
    }
#endif
    // stage B tile 32x128, transposed into Bs[n][k] (manual: transpose in flight)
#pragma unroll
    for (int i = 0; i < 8; ++i) {
      int idx = tid + i * 256;
      int kr = idx >> 6;
      int nc = (idx & 63) << 1;
      uint32_t d = *(const uint32_t*)(B + (size_t)(kt + kr) * N + nb + nc);
      union { uint32_t u; __bf16 b[2]; } t; t.u = d;
      Bs[nc][kr]     = t.b[0];
      Bs[nc + 1][kr] = t.b[1];
    }
#ifdef USE_ASYNC_LDS
    async_wait0();   // A-tile async writes must land before the barrier
#endif
    __syncthreads();

    v16bf afr[4];
#pragma unroll
    for (int mi = 0; mi < 4; ++mi) {
      int m = wm * 64 + mi * 16 + ln;
      afr[mi] = ld8x2(&As[m][8 * lh], &As[m][16 + 8 * lh]);
    }
    v16bf bfr[2];
#pragma unroll
    for (int ni = 0; ni < 2; ++ni) {
      int n = wn * 32 + ni * 16 + ln;
      bfr[ni] = ld8x2(&Bs[n][8 * lh], &Bs[n][16 + 8 * lh]);
    }
#pragma unroll
    for (int mi = 0; mi < 4; ++mi)
#pragma unroll
      for (int ni = 0; ni < 2; ++ni)
        acc[mi][ni] = wmma_bf16(afr[mi], bfr[ni], acc[mi][ni]);
  }

  // epilogue: C layout -> lane holds col n=ln, rows j + 8*lh
#pragma unroll
  for (int mi = 0; mi < 4; ++mi)
#pragma unroll
    for (int ni = 0; ni < 2; ++ni) {
      int n = nb + wn * 32 + ni * 16 + ln;
      float bi = bias ? bias[n] : 0.0f;
#pragma unroll
      for (int j = 0; j < 8; ++j) {
        int m = mb + wm * 64 + mi * 16 + j + 8 * lh;
        float v = acc[mi][ni][j] + bi;
        if (relu) v = fmaxf(v, 0.0f);
        if (resid) v += resid[(size_t)m * N + n];
        if (Cf) Cf[(size_t)m * N + n] = v;
        if (Cb) Cb[(size_t)m * N + n] = (__bf16)v;
      }
    }
}

// ---------------- flash attention (bf16 WMMA, online softmax) ----------------
// grid (S/64, H, B); block = 128 (4 waves); wave owns 16 query rows.
__global__ __launch_bounds__(128)
void flash_attn_kernel(const __bf16* __restrict__ q, const __bf16* __restrict__ k,
                       const __bf16* __restrict__ v, const unsigned char* __restrict__ mask,
                       __bf16* __restrict__ attnb) {
  __shared__ __align__(16) __bf16 Ktile[32][72];      // [key][d]
  __shared__ __align__(16) __bf16 VT[64][40];         // [d][key] (transposed)
  __shared__ __align__(16) __bf16 Pl[4][16][32];      // per-wave P staging

  const int tid  = threadIdx.x;
  const int lane = tid & 31;
  const int ln   = lane & 15;
  const int lh   = lane >> 4;
  const int wave = tid >> 5;
  const int b    = blockIdx.z;
  const int h    = blockIdx.y;
  const int qb0  = blockIdx.x * 64;
  const int headoff = h * DD;

  // Q fragments for this wave's 16 rows (stay in registers for whole K loop)
  const __bf16* qrow = q + (size_t)(b * SS + qb0 + wave * 16 + ln) * EE + headoff;
  v16bf aq[2];
  aq[0] = ld8x2(qrow + 8 * lh,      qrow + 16 + 8 * lh);
  aq[1] = ld8x2(qrow + 32 + 8 * lh, qrow + 48 + 8 * lh);

  const v8f vzero = {0.f, 0.f, 0.f, 0.f, 0.f, 0.f, 0.f, 0.f};
  v8f o[4];
#pragma unroll
  for (int t = 0; t < 4; ++t) o[t] = vzero;
  float mi[8], li[8];
#pragma unroll
  for (int j = 0; j < 8; ++j) { mi[j] = -1.0e30f; li[j] = 0.0f; }

  const __bf16* kbase = k + (size_t)(b * SS) * EE + headoff;
  const __bf16* vbase = v + (size_t)(b * SS) * EE + headoff;

  for (int kb = 0; kb < SS; kb += 32) {
    __syncthreads();
#ifdef USE_ASYNC_LDS
    // stage K tile (straight copy) via async-to-LDS (2 x b128 per lane)
#pragma unroll
    for (int i = 0; i < 2; ++i) {
      int c  = tid + i * 128;          // 256 chunks of 16B
      int kr = c >> 3;                 // 8 chunks per 128B row
      int e  = (c & 7) << 3;
      async_copy16(kbase + (size_t)(kb + kr) * EE + e, &Ktile[kr][e]);
    }
    // stage V transposed (manual)
#pragma unroll
    for (int i = 0; i < 8; ++i) {
      int idx = tid + i * 128;
      int kr = idx >> 5;
      int d  = (idx & 31) << 1;
      uint32_t vv = *(const uint32_t*)(vbase + (size_t)(kb + kr) * EE + d);
      union { uint32_t u; __bf16 b2[2]; } t; t.u = vv;
      VT[d][kr]     = t.b2[0];
      VT[d + 1][kr] = t.b2[1];
    }
    async_wait0();
#else
    // stage K (row-major) and V (transposed): 1024 dwords each / 128 threads
#pragma unroll
    for (int i = 0; i < 8; ++i) {
      int idx = tid + i * 128;
      int kr = idx >> 5;
      int d  = (idx & 31) << 1;
      uint32_t kv = *(const uint32_t*)(kbase + (size_t)(kb + kr) * EE + d);
      *(uint32_t*)&Ktile[kr][d] = kv;
      uint32_t vv = *(const uint32_t*)(vbase + (size_t)(kb + kr) * EE + d);
      union { uint32_t u; __bf16 b2[2]; } t; t.u = vv;
      VT[d][kr]     = t.b2[0];
      VT[d + 1][kr] = t.b2[1];
    }
#endif
    __syncthreads();

    // scores: S[16q,32k] = Q[16,64] @ K^T, two 16x16 C tiles, K-dim = 64
    v8f s[2];
#pragma unroll
    for (int g = 0; g < 2; ++g) {
      v8f a = vzero;
#pragma unroll
      for (int dh = 0; dh < 2; ++dh) {
        v16bf bk = ld8x2(&Ktile[16 * g + ln][32 * dh + 8 * lh], &Ktile[16 * g + ln][32 * dh + 16 + 8 * lh]);
        a = wmma_bf16(aq[dh], bk, a);
      }
      s[g] = a;
    }
    // scale + key mask bias (lane's column = one key for all 8 rows)
#pragma unroll
    for (int g = 0; g < 2; ++g) {
      int key = kb + 16 * g + ln;
      float mb = mask[b * SS + key] ? 0.0f : -1.0e9f;
#pragma unroll
      for (int j = 0; j < 8; ++j) s[g][j] = s[g][j] * 0.125f + mb;
    }
    // online softmax: row stats live in the 16 lanes of each half-wave
    float tm[8];
#pragma unroll
    for (int j = 0; j < 8; ++j) tm[j] = fmaxf(s[0][j], s[1][j]);
#pragma unroll
    for (int off = 1; off < 16; off <<= 1)
#pragma unroll
      for (int j = 0; j < 8; ++j) tm[j] = fmaxf(tm[j], __shfl_xor(tm[j], off, 32));
    float ts[8];
#pragma unroll
    for (int j = 0; j < 8; ++j) {
      float nm = fmaxf(mi[j], tm[j]);
      float corr = __expf(mi[j] - nm);
      s[0][j] = __expf(s[0][j] - nm);
      s[1][j] = __expf(s[1][j] - nm);
      ts[j] = s[0][j] + s[1][j];
      li[j] *= corr;
      mi[j] = nm;
#pragma unroll
      for (int t = 0; t < 4; ++t) o[t][j] *= corr;
    }
#pragma unroll
    for (int off = 1; off < 16; off <<= 1)
#pragma unroll
      for (int j = 0; j < 8; ++j) ts[j] += __shfl_xor(ts[j], off, 32);
#pragma unroll
    for (int j = 0; j < 8; ++j) li[j] += ts[j];

    // C-layout P -> LDS -> A-layout fragment (wave-private region, DS in-order)
#pragma unroll
    for (int g = 0; g < 2; ++g)
#pragma unroll
      for (int j = 0; j < 8; ++j)
        Pl[wave][j + 8 * lh][16 * g + ln] = (__bf16)s[g][j];
    v16bf pa = ld8x2(&Pl[wave][ln][8 * lh], &Pl[wave][ln][16 + 8 * lh]);

    // O[16,64] += P[16,32] @ V[32,64]
#pragma unroll
    for (int t = 0; t < 4; ++t) {
      v16bf bv = ld8x2(&VT[16 * t + ln][8 * lh], &VT[16 * t + ln][16 + 8 * lh]);
      o[t] = wmma_bf16(pa, bv, o[t]);
    }
  }

  // normalize and emit bf16 attention output
  float inv[8];
#pragma unroll
  for (int j = 0; j < 8; ++j) inv[j] = 1.0f / fmaxf(li[j], 1.0e-30f);
#pragma unroll
  for (int t = 0; t < 4; ++t)
#pragma unroll
    for (int j = 0; j < 8; ++j) {
      int row = b * SS + qb0 + wave * 16 + j + 8 * lh;
      int col = headoff + 16 * t + ln;
      attnb[(size_t)row * EE + col] = (__bf16)(o[t][j] * inv[j]);
    }
}

// ---------------- layernorm (eps = 1e-3), optional bf16 copy ----------------
__global__ __launch_bounds__(256)
void layernorm_kernel(const float* __restrict__ x, const float* __restrict__ g,
                      const float* __restrict__ bta, float* __restrict__ outf,
                      __bf16* __restrict__ outb) {
  const int row = blockIdx.x;
  const float* xr = x + (size_t)row * EE;
  float s = 0.0f, ss = 0.0f;
  for (int c = threadIdx.x; c < EE; c += 256) {
    float v = xr[c]; s += v; ss += v * v;
  }
  for (int off = 16; off; off >>= 1) {
    s  += __shfl_xor(s, off, 32);
    ss += __shfl_xor(ss, off, 32);
  }
  __shared__ float sh[2][8];
  int w = threadIdx.x >> 5, l = threadIdx.x & 31;
  if (l == 0) { sh[0][w] = s; sh[1][w] = ss; }
  __syncthreads();
  if (w == 0) {
    s  = (l < 8) ? sh[0][l] : 0.0f;
    ss = (l < 8) ? sh[1][l] : 0.0f;
    for (int off = 4; off; off >>= 1) {
      s  += __shfl_xor(s, off, 32);
      ss += __shfl_xor(ss, off, 32);
    }
    if (l == 0) { sh[0][0] = s; sh[1][0] = ss; }
  }
  __syncthreads();
  float mu  = sh[0][0] * (1.0f / EE);
  float var = sh[1][0] * (1.0f / EE) - mu * mu;
  float rs  = rsqrtf(var + 1.0e-3f);
  for (int c = threadIdx.x; c < EE; c += 256) {
    float y = (xr[c] - mu) * rs * g[c] + bta[c];
    outf[(size_t)row * EE + c] = y;
    if (outb) outb[(size_t)row * EE + c] = (__bf16)y;
  }
}

// ---------------- host-side orchestration ----------------
extern "C" void kernel_launch(void* const* d_in, const int* in_sizes, int n_in,
                              void* d_out, int out_size, void* d_ws, size_t ws_size,
                              hipStream_t stream) {
  const float*         x    = (const float*)d_in[0];
  const unsigned char* mask = (const unsigned char*)d_in[1];
  const float* Wq = (const float*)d_in[2];  const float* bq = (const float*)d_in[3];
  const float* Wk = (const float*)d_in[4];  const float* bk = (const float*)d_in[5];
  const float* Wv = (const float*)d_in[6];  const float* bv = (const float*)d_in[7];
  const float* Wo = (const float*)d_in[8];  const float* bo = (const float*)d_in[9];
  const float* W1 = (const float*)d_in[10]; const float* b1 = (const float*)d_in[11];
  const float* W2 = (const float*)d_in[12]; const float* b2 = (const float*)d_in[13];
  const float* ln1g = (const float*)d_in[14]; const float* ln1b = (const float*)d_in[15];
  const float* ln2g = (const float*)d_in[16]; const float* ln2b = (const float*)d_in[17];

  char* ws = (char*)d_ws;
  auto alloc = [&](size_t bytes) -> void* {
    void* p = ws;
    ws += (bytes + 255) & ~(size_t)255;
    return p;
  };
  const size_t ME = (size_t)MM * EE;
  __bf16* xb   = (__bf16*)alloc(ME * 2);
  __bf16* wqb  = (__bf16*)alloc((size_t)EE * EE * 2);
  __bf16* wkb  = (__bf16*)alloc((size_t)EE * EE * 2);
  __bf16* wvb  = (__bf16*)alloc((size_t)EE * EE * 2);
  __bf16* wob  = (__bf16*)alloc((size_t)EE * EE * 2);
  __bf16* w1b  = (__bf16*)alloc((size_t)EE * FF_ * 2);
  __bf16* w2b  = (__bf16*)alloc((size_t)FF_ * EE * 2);
  __bf16* qb   = (__bf16*)alloc(ME * 2);
  __bf16* kb_  = (__bf16*)alloc(ME * 2);
  __bf16* vb_  = (__bf16*)alloc(ME * 2);
  __bf16* attb = (__bf16*)alloc(ME * 2);
  float*  y1   = (float*)alloc(ME * 4);
  float*  h32  = (float*)alloc(ME * 4);
  __bf16* hb   = (__bf16*)alloc(ME * 2);
  __bf16* ff1b = (__bf16*)alloc((size_t)MM * FF_ * 2);
  float*  y2   = (float*)alloc(ME * 4);

  // fp32 -> bf16 conversions
  cvt_bf16_kernel<<<2048, 256, 0, stream>>>(x,  xb,  (int)ME);
  cvt_bf16_kernel<<<512,  256, 0, stream>>>(Wq, wqb, EE * EE);
  cvt_bf16_kernel<<<512,  256, 0, stream>>>(Wk, wkb, EE * EE);
  cvt_bf16_kernel<<<512,  256, 0, stream>>>(Wv, wvb, EE * EE);
  cvt_bf16_kernel<<<512,  256, 0, stream>>>(Wo, wob, EE * EE);
  cvt_bf16_kernel<<<2048, 256, 0, stream>>>(W1, w1b, EE * FF_);
  cvt_bf16_kernel<<<2048, 256, 0, stream>>>(W2, w2b, FF_ * EE);

  dim3 gE(EE / 128, MM / 128);    // N=1024 GEMMs
  dim3 gF(FF_ / 128, MM / 128);   // N=4096 GEMM

  // QKV projections (bf16 out)
  gemm_bf16_kernel<<<gE, 256, 0, stream>>>(xb, wqb, bq, nullptr, nullptr, qb,  MM, EE, EE, 0);
  gemm_bf16_kernel<<<gE, 256, 0, stream>>>(xb, wkb, bk, nullptr, nullptr, kb_, MM, EE, EE, 0);
  gemm_bf16_kernel<<<gE, 256, 0, stream>>>(xb, wvb, bv, nullptr, nullptr, vb_, MM, EE, EE, 0);

  // flash attention -> bf16 context
  flash_attn_kernel<<<dim3(SS / 64, HH, BB), 128, 0, stream>>>(qb, kb_, vb_, mask, attb);

  // output projection + residual(x) -> y1; LN1 -> h (f32 + bf16)
  gemm_bf16_kernel<<<gE, 256, 0, stream>>>(attb, wob, bo, x, y1, nullptr, MM, EE, EE, 0);
  layernorm_kernel<<<MM, 256, 0, stream>>>(y1, ln1g, ln1b, h32, hb);

  // FFN: relu(h@W1+b1) -> ff1 (bf16); ff1@W2+b2 + h -> y2; LN2 -> out
  gemm_bf16_kernel<<<gF, 256, 0, stream>>>(hb, w1b, b1, nullptr, nullptr, ff1b, MM, FF_, EE, 1);
  gemm_bf16_kernel<<<gE, 256, 0, stream>>>(ff1b, w2b, b2, h32, y2, nullptr, MM, EE, FF_, 0);
  layernorm_kernel<<<MM, 256, 0, stream>>>(y2, ln2g, ln2b, (float*)d_out, nullptr);
}
